// SAGE_3590592659701
// MI455X (gfx1250) — matrix operand order repaired and checked by reading, not compiled, not run on hardware
//
#include <hip/hip_runtime.h>

typedef __attribute__((ext_vector_type(16))) _Float16 v16h;
typedef __attribute__((ext_vector_type(8)))  _Float16 v8h;
typedef __attribute__((ext_vector_type(8)))  float    v8f;

#define D 128
#define TILE_M 16
#define LDS_STRIDE (D + 8)

union AFrag { v16h v; v8h h[2]; };

// ---------------- utility kernels ----------------

__global__ void fill_zero(float* __restrict__ p, int n) {
  int t = blockIdx.x * blockDim.x + threadIdx.x;
  if (t < n) p[t] = 0.0f;
}

__global__ void count_deg(const int* __restrict__ dst, float* __restrict__ deg, int E) {
  int t = blockIdx.x * blockDim.x + threadIdx.x;
  if (t < E) atomicAdd(&deg[dst[t]], 1.0f);
}

__global__ void make_rdeg(const float* __restrict__ deg, float* __restrict__ rdeg, int N) {
  int t = blockIdx.x * blockDim.x + threadIdx.x;
  if (t < N) rdeg[t] = 1.0f / fmaxf(deg[t], 1.0f);
}

// Convert 8 [128x128] fp32 matrices to transposed f16: Wt[m][n][k] = W[m][k][n]
// m = 0..2 : W_self[l], m = 3..5 : W_neigh[l], m = 6 : Wp1, m = 7 : Wp2
__global__ void convert_weights(const float* __restrict__ Wself,
                                const float* __restrict__ Wneigh,
                                const float* __restrict__ Wp1,
                                const float* __restrict__ Wp2,
                                _Float16* __restrict__ Wt) {
  int t = blockIdx.x * blockDim.x + threadIdx.x;
  if (t >= 8 * D * D) return;
  int m = t >> 14;
  int r = t & 16383;
  int n = r >> 7;
  int k = r & 127;
  float v;
  if (m < 3)       v = Wself [(size_t)m * D * D + k * D + n];
  else if (m < 6)  v = Wneigh[(size_t)(m - 3) * D * D + k * D + n];
  else if (m == 6) v = Wp1[k * D + n];
  else             v = Wp2[k * D + n];
  Wt[(size_t)m * D * D + n * D + k] = (_Float16)v;
}

// Scatter-add: agg[dst[e]] += h[src[e]], one thread per (edge, 4 floats).
// h rows are 512B-contiguous -> coalesced gather; atomics resolve at L2.
__global__ void scatter_add(const float* __restrict__ h,
                            const int* __restrict__ src,
                            const int* __restrict__ dst,
                            float* __restrict__ agg, int E) {
  int t = blockIdx.x * blockDim.x + threadIdx.x;
  int e = t >> 5;
  if (e >= E) return;
  int c = (t & 31) * 4;
  const float4 v = *(const float4*)&h[(size_t)src[e] * D + c];
  float* a = &agg[(size_t)dst[e] * D + c];
  atomicAdd(a + 0, v.x);
  atomicAdd(a + 1, v.y);
  atomicAdd(a + 2, v.z);
  atomicAdd(a + 3, v.w);
}

// ---------------- fused SAGE layer: h_out = act(h@Ws + (agg/deg)@Wn + b) ----------------
// 256 threads = 8 waves; block owns 16 nodes; wave w owns output cols [16w,16w+16).
__global__ __launch_bounds__(256) void sage_layer(
    const float* __restrict__ hin, const float* __restrict__ agg,
    const float* __restrict__ rdeg,
    const _Float16* __restrict__ WtS, const _Float16* __restrict__ WtN,
    const float* __restrict__ bias, float* __restrict__ hout, int do_relu) {
  __shared__ _Float16 As[TILE_M][LDS_STRIDE];
  __shared__ _Float16 Ag[TILE_M][LDS_STRIDE];
  const int t = threadIdx.x;
  const int node0 = blockIdx.x * TILE_M;

  // stage: each thread converts 8 contiguous f32 of h and agg into f16 LDS tiles
  {
    int idx = t * 8;
    int row = idx >> 7;
    int col = idx & 127;
    const float4* hp = (const float4*)&hin[(size_t)(node0 + row) * D + col];
    const float4* ap = (const float4*)&agg[(size_t)(node0 + row) * D + col];
    float4 h0 = hp[0], h1 = hp[1];
    float4 a0 = ap[0], a1 = ap[1];
    float rd = rdeg[node0 + row];
    _Float16* os = &As[row][col];
    os[0] = (_Float16)h0.x; os[1] = (_Float16)h0.y;
    os[2] = (_Float16)h0.z; os[3] = (_Float16)h0.w;
    os[4] = (_Float16)h1.x; os[5] = (_Float16)h1.y;
    os[6] = (_Float16)h1.z; os[7] = (_Float16)h1.w;
    _Float16* oa = &Ag[row][col];
    oa[0] = (_Float16)(a0.x * rd); oa[1] = (_Float16)(a0.y * rd);
    oa[2] = (_Float16)(a0.z * rd); oa[3] = (_Float16)(a0.w * rd);
    oa[4] = (_Float16)(a1.x * rd); oa[5] = (_Float16)(a1.y * rd);
    oa[6] = (_Float16)(a1.z * rd); oa[7] = (_Float16)(a1.w * rd);
  }
  __syncthreads();

  const int lane = t & 31;
  const int wv   = t >> 5;
  const int nloc = lane & 15;
  const int hi   = lane >> 4;
  const int kb   = hi * 16;   // A-fragment K sub-offset: contiguous k = 16*hi + s
  const int koff = hi * 16;   // B-fragment K sub-offset: contiguous k = 16*hi + s, same as A
  const int col  = wv * 16 + nloc;

  v8f acc;
  const float bv = bias[col];
  for (int r = 0; r < 8; ++r) acc[r] = bv;

  const _Float16* bs = WtS + (size_t)col * D;  // transposed: 16 contiguous K per lane
  const _Float16* bn = WtN + (size_t)col * D;

#pragma unroll
  for (int k0 = 0; k0 < D; k0 += 32) {
    AFrag a;
    a.h[0] = *(const v8h*)&As[nloc][k0 + kb];
    a.h[1] = *(const v8h*)&As[nloc][k0 + kb + 8];
    v16h b0 = *(const v16h*)&bs[k0 + koff];
    acc = __builtin_amdgcn_wmma_f32_16x16x32_f16(false, a.v, false, b0,
                                                 (short)0, acc, false, false);
    a.h[0] = *(const v8h*)&Ag[nloc][k0 + kb];
    a.h[1] = *(const v8h*)&Ag[nloc][k0 + kb + 8];
    v16h b1 = *(const v16h*)&bn[k0 + koff];
    acc = __builtin_amdgcn_wmma_f32_16x16x32_f16(false, a.v, false, b1,
                                                 (short)0, acc, false, false);
  }

#pragma unroll
  for (int r = 0; r < 8; ++r) {
    float v = acc[r];
    if (do_relu) v = fmaxf(v, 0.0f);
    int row = hi * 8 + r;  // C/D f32 VGPR layout
    hout[(size_t)(node0 + row) * D + col] = v;
  }
}

// ---------------- fully fused 3-layer MLP predictor on 16 edges per block ----------------
__global__ __launch_bounds__(256) void predictor16(
    const float* __restrict__ h,
    const int* __restrict__ src, const int* __restrict__ dst,
    const _Float16* __restrict__ Wt1, const float* __restrict__ bp1,
    const _Float16* __restrict__ Wt2, const float* __restrict__ bp2,
    const float* __restrict__ Wp3, const float* __restrict__ bp3,
    float* __restrict__ out) {
  __shared__ _Float16 T0[TILE_M][LDS_STRIDE];
  __shared__ _Float16 T1[TILE_M][LDS_STRIDE];
  __shared__ int sS[TILE_M], sD[TILE_M];
  const int t = threadIdx.x;
  const int e0 = blockIdx.x * TILE_M;

  if (t < TILE_M) { sS[t] = src[e0 + t]; sD[t] = dst[e0 + t]; }
  __syncthreads();

  // stage elementwise product h[src]*h[dst] as f16
  {
    int idx = t * 8;
    int row = idx >> 7;
    int col = idx & 127;
    const float4* sp = (const float4*)&h[(size_t)sS[row] * D + col];
    const float4* dp = (const float4*)&h[(size_t)sD[row] * D + col];
    float4 s0 = sp[0], s1 = sp[1], d0 = dp[0], d1 = dp[1];
    _Float16* o = &T0[row][col];
    o[0] = (_Float16)(s0.x * d0.x); o[1] = (_Float16)(s0.y * d0.y);
    o[2] = (_Float16)(s0.z * d0.z); o[3] = (_Float16)(s0.w * d0.w);
    o[4] = (_Float16)(s1.x * d1.x); o[5] = (_Float16)(s1.y * d1.y);
    o[6] = (_Float16)(s1.z * d1.z); o[7] = (_Float16)(s1.w * d1.w);
  }
  __syncthreads();

  const int lane = t & 31;
  const int wv   = t >> 5;
  const int nloc = lane & 15;
  const int hi   = lane >> 4;
  const int kb   = hi * 16;
  const int koff = hi * 16;
  const int col  = wv * 16 + nloc;

  // layer 1: T1 = relu(T0 @ Wp1 + bp1)
  {
    v8f acc;
    const float bv = bp1[col];
    for (int r = 0; r < 8; ++r) acc[r] = bv;
    const _Float16* bw = Wt1 + (size_t)col * D;
#pragma unroll
    for (int k0 = 0; k0 < D; k0 += 32) {
      AFrag a;
      a.h[0] = *(const v8h*)&T0[nloc][k0 + kb];
      a.h[1] = *(const v8h*)&T0[nloc][k0 + kb + 8];
      v16h b = *(const v16h*)&bw[k0 + koff];
      acc = __builtin_amdgcn_wmma_f32_16x16x32_f16(false, a.v, false, b,
                                                   (short)0, acc, false, false);
    }
#pragma unroll
    for (int r = 0; r < 8; ++r)
      T1[hi * 8 + r][col] = (_Float16)fmaxf(acc[r], 0.0f);
  }
  __syncthreads();

  // layer 2: T0 = relu(T1 @ Wp2 + bp2)
  {
    v8f acc;
    const float bv = bp2[col];
    for (int r = 0; r < 8; ++r) acc[r] = bv;
    const _Float16* bw = Wt2 + (size_t)col * D;
#pragma unroll
    for (int k0 = 0; k0 < D; k0 += 32) {
      AFrag a;
      a.h[0] = *(const v8h*)&T1[nloc][k0 + kb];
      a.h[1] = *(const v8h*)&T1[nloc][k0 + kb + 8];
      v16h b = *(const v16h*)&bw[k0 + koff];
      acc = __builtin_amdgcn_wmma_f32_16x16x32_f16(false, a.v, false, b,
                                                   (short)0, acc, false, false);
    }
#pragma unroll
    for (int r = 0; r < 8; ++r)
      T0[hi * 8 + r][col] = (_Float16)fmaxf(acc[r], 0.0f);
  }
  __syncthreads();

  // layer 3: [16,128] @ [128,1] + bp3 — tiny, plain VALU dot
  if (t < TILE_M) {
    float s = bp3[0];
#pragma unroll 8
    for (int k = 0; k < D; ++k) s += (float)T0[t][k] * Wp3[k];
    out[e0 + t] = s;
  }
}

// ---------------- host launch ----------------

extern "C" void kernel_launch(void* const* d_in, const int* in_sizes, int n_in,
                              void* d_out, int out_size, void* d_ws, size_t ws_size,
                              hipStream_t stream) {
  (void)n_in; (void)out_size; (void)ws_size;
  const float* x      = (const float*)d_in[0];
  const float* Wself  = (const float*)d_in[1];
  const float* Wneigh = (const float*)d_in[2];
  const float* bvec   = (const float*)d_in[3];
  const float* Wp1    = (const float*)d_in[4];
  const float* bp1    = (const float*)d_in[5];
  const float* Wp2    = (const float*)d_in[6];
  const float* bp2    = (const float*)d_in[7];
  const float* Wp3    = (const float*)d_in[8];
  const float* bp3    = (const float*)d_in[9];
  const int* esrc = (const int*)d_in[10];
  const int* edst = (const int*)d_in[11];
  const int* psrc = (const int*)d_in[12];
  const int* pdst = (const int*)d_in[13];
  const int* nsrc = (const int*)d_in[14];
  const int* ndst = (const int*)d_in[15];

  const int N = in_sizes[0] / D;
  const int E = in_sizes[10];
  const int P = in_sizes[12];
  float* out = (float*)d_out;

  // workspace carve-out (256B aligned)
  char* ws = (char*)d_ws;
  size_t off = 0;
  auto alloc = [&](size_t bytes) -> char* {
    char* p = ws + off;
    off = (off + bytes + 255) & ~(size_t)255;
    return p;
  };
  float*    deg  = (float*)alloc((size_t)N * 4);
  float*    rdeg = (float*)alloc((size_t)N * 4);
  float*    hA   = (float*)alloc((size_t)N * D * 4);
  float*    hB   = (float*)alloc((size_t)N * D * 4);
  float*    agg  = (float*)alloc((size_t)N * D * 4);
  _Float16* Wt   = (_Float16*)alloc((size_t)8 * D * D * 2);

  const int TB = 256;
  fill_zero<<<(N + TB - 1) / TB, TB, 0, stream>>>(deg, N);
  convert_weights<<<(8 * D * D + TB - 1) / TB, TB, 0, stream>>>(Wself, Wneigh, Wp1, Wp2, Wt);
  count_deg<<<(E + TB - 1) / TB, TB, 0, stream>>>(edst, deg, E);
  make_rdeg<<<(N + TB - 1) / TB, TB, 0, stream>>>(deg, rdeg, N);

  const float* cur = x;
  for (int l = 0; l < 3; ++l) {
    float* nxt = (l == 0) ? hA : ((l == 1) ? hB : hA);
    int nElem = N * D;
    fill_zero<<<(nElem + TB - 1) / TB, TB, 0, stream>>>(agg, nElem);
    long sw = (long)E * 32;
    scatter_add<<<(int)((sw + TB - 1) / TB), TB, 0, stream>>>(cur, esrc, edst, agg, E);
    sage_layer<<<N / TILE_M, TB, 0, stream>>>(
        cur, agg, rdeg,
        Wt + (size_t)l * D * D, Wt + (size_t)(3 + l) * D * D,
        bvec + l * D, nxt, (l < 2) ? 1 : 0);
    cur = nxt;
  }

  predictor16<<<P / TILE_M, TB, 0, stream>>>(
      cur, psrc, pdst, Wt + (size_t)6 * D * D, bp1,
      Wt + (size_t)7 * D * D, bp2, Wp3, bp3, out);
  predictor16<<<P / TILE_M, TB, 0, stream>>>(
      cur, nsrc, ndst, Wt + (size_t)6 * D * D, bp1,
      Wt + (size_t)7 * D * D, bp2, Wp3, bp3, out + P);
}
